// SCDCLCELoss_8761733284020
// MI455X (gfx1250) — matrix operand order repaired and checked by reading, not compiled, not run on hardware
//
#include <hip/hip_runtime.h>

// ---------------------------------------------------------------------------
// Supervised contrastive loss, fused, for MI455X (gfx1250, wave32, WMMA).
//   N = 4096 rows (view-major reorder of feats[A=256, V=16, D=256])
//   logits = X X^T / 0.07  via bf16 hi/lo split-3 WMMA (≈ f32 accurate)
//   pass1: per-row online (max, neg_sum);  pass2: positive log-prob sums.
// Round-1 fix: rolled K-loop (unroll 2) -> no scratch spills.
// Round-2 fix: 32-bit index addressing off kernel-arg bases (no PHI-carried
//   pointers) so loads lower to GLOBAL (saddr+voffset), not FLAT.
// ---------------------------------------------------------------------------

typedef __attribute__((ext_vector_type(16))) __bf16 v16bf;
typedef __attribute__((ext_vector_type(8)))  float  v8f;

#define AN   256           // anchors
#define NV   16            // views
#define NT   4096          // N = AN*NV
#define DD   256           // feature dim
#define INVT (1.0f / 0.07f)

union Frag { v16bf v; unsigned int u[8]; };

__device__ __forceinline__ unsigned short f2bf_rne(float f) {
  unsigned int u = __float_as_uint(f);
  u += 0x7fffu + ((u >> 16) & 1u);          // round-to-nearest-even
  return (unsigned short)(u >> 16);
}
__device__ __forceinline__ float bf2f(unsigned short h) {
  return __uint_as_float(((unsigned int)h) << 16);
}

// ---------------------------------------------------------------------------
// Kernel 0: reorder [A,V,D] -> view-major [N,D], split f32 into bf16 hi+lo.
// Also zero the scalar output (deterministic across graph replays).
// ---------------------------------------------------------------------------
__global__ void k_convert(const float* __restrict__ feats,
                          unsigned short* __restrict__ Xhi,
                          unsigned short* __restrict__ Xlo,
                          float* __restrict__ out) {
  int idx = blockIdx.x * blockDim.x + threadIdx.x;   // over NT*DD
  if (idx == 0) *out = 0.0f;
  if (idx >= NT * DD) return;
  int n = idx >> 8;            // row in contrast ordering
  int d = idx & (DD - 1);
  int a = n & (AN - 1);        // n = v*AN + a
  int v = n >> 8;
  float x = feats[(size_t)(a * NV + v) * DD + d];
  unsigned short hi = f2bf_rne(x);
  float res = x - bf2f(hi);
  Xhi[idx] = hi;
  Xlo[idx] = f2bf_rne(res);
}

// ---------------------------------------------------------------------------
// 32x32 output tile (2x2 WMMA tiles) of X X^T using bf16 split-3.
// A layout (16-bit 16x32): lane m=L&15, u[v] -> K pairs {2v,2v+1}+16h,
// i.e. K=16h+i.  B layout (same order): lane n=L&15, elems i -> K=16h+i.
// D layout: lane L -> col = L&15, VGPR r -> row = r + 8*(L>>4).
// Rolled K-loop (unroll 2) bounds VGPR pressure; 32-bit indices off the
// kernel-arg bases keep loads on the GLOBAL path.
// ---------------------------------------------------------------------------
__device__ __forceinline__ void gemm32x32(const unsigned short* __restrict__ Xhi,
                                          const unsigned short* __restrict__ Xlo,
                                          int row0, int col0, v8f acc[2][2]) {
  const int lane = threadIdx.x & 31;
  const int l15  = lane & 15;
  const int h    = lane >> 4;

#pragma unroll
  for (int i = 0; i < 2; ++i)
#pragma unroll
    for (int j = 0; j < 2; ++j)
#pragma unroll
      for (int r = 0; r < 8; ++r) acc[i][j][r] = 0.0f;

  const int aoff0 = (row0 + l15) * DD;            // + tr*16*DD + k0 + kb
  const int boff0 = (col0 + l15) * DD + h * 16;   // + tc*16*DD + k0 + 2v

#pragma unroll 2
  for (int k0 = 0; k0 < DD; k0 += 32) {
    Frag ahi[2], alo[2], bhi[2], blo[2];
#pragma unroll
    for (int tr = 0; tr < 2; ++tr) {
      const int base = aoff0 + tr * 16 * DD + k0;
#pragma unroll
      for (int v = 0; v < 8; ++v) {
        const int kb = (h << 4) + (v << 1);
        ahi[tr].u[v] = *(const unsigned int*)(Xhi + base + kb);
        alo[tr].u[v] = *(const unsigned int*)(Xlo + base + kb);
      }
    }
#pragma unroll
    for (int tc = 0; tc < 2; ++tc) {
      const int base = boff0 + tc * 16 * DD + k0;
#pragma unroll
      for (int v = 0; v < 8; ++v) {
        bhi[tc].u[v] = *(const unsigned int*)(Xhi + base + 2 * v);
        blo[tc].u[v] = *(const unsigned int*)(Xlo + base + 2 * v);
      }
    }
#pragma unroll
    for (int tr = 0; tr < 2; ++tr)
#pragma unroll
      for (int tc = 0; tc < 2; ++tc) {
        acc[tr][tc] = __builtin_amdgcn_wmma_f32_16x16x32_bf16(
            false, ahi[tr].v, false, bhi[tc].v, (short)0, acc[tr][tc], false, false);
        acc[tr][tc] = __builtin_amdgcn_wmma_f32_16x16x32_bf16(
            false, ahi[tr].v, false, blo[tc].v, (short)0, acc[tr][tc], false, false);
        acc[tr][tc] = __builtin_amdgcn_wmma_f32_16x16x32_bf16(
            false, alo[tr].v, false, bhi[tc].v, (short)0, acc[tr][tc], false, false);
      }
  }
}

// ---------------------------------------------------------------------------
// Pass 1: per-row max over full row and neg_sum = sum_{lbl_j != lbl_i} e^(l-m)
// Block = 8 waves, 32-row stripe; each wave sweeps 16 column blocks of 32.
// ---------------------------------------------------------------------------
__global__ void k_stats(const unsigned short* __restrict__ Xhi,
                        const unsigned short* __restrict__ Xlo,
                        const int* __restrict__ labels,
                        float* __restrict__ mrow, float* __restrict__ srow) {
  __shared__ float sm[8][32];
  __shared__ float ss[8][32];
  const int row0 = blockIdx.x * 32;
  const int w    = threadIdx.x >> 5;
  const int lane = threadIdx.x & 31;
  const int l15  = lane & 15;
  const int h    = lane >> 4;

  int lbli[2][8];
#pragma unroll
  for (int tr = 0; tr < 2; ++tr)
#pragma unroll
    for (int r = 0; r < 8; ++r)
      lbli[tr][r] = labels[(row0 + tr * 16 + h * 8 + r) & (AN - 1)];

  float mr[2][8], sr[2][8];
#pragma unroll
  for (int tr = 0; tr < 2; ++tr)
#pragma unroll
    for (int r = 0; r < 8; ++r) { mr[tr][r] = -1.0e30f; sr[tr][r] = 0.0f; }

  for (int cb = 0; cb < 16; ++cb) {
    const int col0 = (cb * 8 + w) * 32;
    v8f acc[2][2];
    gemm32x32(Xhi, Xlo, row0, col0, acc);
#pragma unroll
    for (int tc = 0; tc < 2; ++tc) {
      const int lblj = labels[(col0 + tc * 16 + l15) & (AN - 1)];
#pragma unroll
      for (int tr = 0; tr < 2; ++tr)
#pragma unroll
        for (int r = 0; r < 8; ++r) {
          const float l  = acc[tr][tc][r] * INVT;
          const float nm = fmaxf(mr[tr][r], l);
          if (nm > mr[tr][r]) { sr[tr][r] *= __expf(mr[tr][r] - nm); mr[tr][r] = nm; }
          if (lblj != lbli[tr][r]) sr[tr][r] += __expf(l - nm);
        }
    }
  }
  // merge (m,s) across the 16 lanes holding the same row
#pragma unroll
  for (int off = 1; off < 16; off <<= 1)
#pragma unroll
    for (int tr = 0; tr < 2; ++tr)
#pragma unroll
      for (int r = 0; r < 8; ++r) {
        const float om = __shfl_xor(mr[tr][r], off, 32);
        const float os = __shfl_xor(sr[tr][r], off, 32);
        const float nm = fmaxf(mr[tr][r], om);
        sr[tr][r] = sr[tr][r] * __expf(mr[tr][r] - nm) + os * __expf(om - nm);
        mr[tr][r] = nm;
      }
  if (l15 == 0) {
#pragma unroll
    for (int tr = 0; tr < 2; ++tr)
#pragma unroll
      for (int r = 0; r < 8; ++r) {
        const int rl = tr * 16 + h * 8 + r;
        sm[w][rl] = mr[tr][r];
        ss[w][rl] = sr[tr][r];
      }
  }
  __syncthreads();
  if (threadIdx.x < 32) {
    float m = sm[0][threadIdx.x], s = ss[0][threadIdx.x];
#pragma unroll
    for (int wv = 1; wv < 8; ++wv) {
      const float om = sm[wv][threadIdx.x], os = ss[wv][threadIdx.x];
      const float nm = fmaxf(m, om);
      s = s * __expf(m - nm) + os * __expf(om - nm);
      m = nm;
    }
    mrow[row0 + threadIdx.x] = m;
    srow[row0 + threadIdx.x] = s;
  }
}

// ---------------------------------------------------------------------------
// Pass 2: sum over positives of (l-m) - log(e^(l-m) + neg_sum); finalize loss.
// ---------------------------------------------------------------------------
__global__ void k_loss(const unsigned short* __restrict__ Xhi,
                       const unsigned short* __restrict__ Xlo,
                       const int* __restrict__ labels,
                       const float* __restrict__ mrow,
                       const float* __restrict__ srow,
                       float* __restrict__ out) {
  __shared__ float sp[8][32];
  const int row0 = blockIdx.x * 32;
  const int w    = threadIdx.x >> 5;
  const int lane = threadIdx.x & 31;
  const int l15  = lane & 15;
  const int h    = lane >> 4;

  int   lbli[2][8];
  float mi[2][8], si[2][8], ps[2][8];
#pragma unroll
  for (int tr = 0; tr < 2; ++tr)
#pragma unroll
    for (int r = 0; r < 8; ++r) {
      const int row = row0 + tr * 16 + h * 8 + r;
      lbli[tr][r] = labels[row & (AN - 1)];
      mi[tr][r]   = mrow[row];
      si[tr][r]   = srow[row];
      ps[tr][r]   = 0.0f;
    }

  for (int cb = 0; cb < 16; ++cb) {
    const int col0 = (cb * 8 + w) * 32;
    v8f acc[2][2];
    gemm32x32(Xhi, Xlo, row0, col0, acc);
#pragma unroll
    for (int tc = 0; tc < 2; ++tc) {
      const int j    = col0 + tc * 16 + l15;
      const int lblj = labels[j & (AN - 1)];
#pragma unroll
      for (int tr = 0; tr < 2; ++tr)
#pragma unroll
        for (int r = 0; r < 8; ++r) {
          const int i = row0 + tr * 16 + h * 8 + r;
          if (lblj == lbli[tr][r] && j != i) {
            const float sh = acc[tr][tc][r] * INVT - mi[tr][r];
            ps[tr][r] += sh - __logf(__expf(sh) + si[tr][r]);
          }
        }
    }
  }
#pragma unroll
  for (int off = 1; off < 16; off <<= 1)
#pragma unroll
    for (int tr = 0; tr < 2; ++tr)
#pragma unroll
      for (int r = 0; r < 8; ++r)
        ps[tr][r] += __shfl_xor(ps[tr][r], off, 32);
  if (l15 == 0) {
#pragma unroll
    for (int tr = 0; tr < 2; ++tr)
#pragma unroll
      for (int r = 0; r < 8; ++r) sp[w][tr * 16 + h * 8 + r] = ps[tr][r];
  }
  __syncthreads();
  if (threadIdx.x < 32) {
    float t = 0.0f;
#pragma unroll
    for (int wv = 0; wv < 8; ++wv) t += sp[wv][threadIdx.x];
    const int row = row0 + threadIdx.x;
    const int lbl = labels[row & (AN - 1)];
    int cnt = 0;
    for (int a = 0; a < AN; ++a) cnt += (labels[a] == lbl) ? 1 : 0;
    const float posN = (float)(cnt * NV - 1);              // always >= 15
    // (TEMP/BASE_TEMP) == 1.0 ; mean over N rows
    float contrib = -(t / posN) / (float)NT;
#pragma unroll
    for (int off = 1; off < 32; off <<= 1) contrib += __shfl_xor(contrib, off, 32);
    if (threadIdx.x == 0) atomicAdd(out, contrib);
  }
}

// ---------------------------------------------------------------------------
extern "C" void kernel_launch(void* const* d_in, const int* in_sizes, int n_in,
                              void* d_out, int out_size, void* d_ws, size_t ws_size,
                              hipStream_t stream) {
  (void)in_sizes; (void)n_in; (void)out_size; (void)ws_size;
  const float* feats  = (const float*)d_in[0];
  const int*   labels = (const int*)d_in[1];
  float*       out    = (float*)d_out;

  char* ws = (char*)d_ws;                       // ~4.03 MB used
  unsigned short* Xhi = (unsigned short*)ws;                               // 2 MB
  unsigned short* Xlo = (unsigned short*)(ws + (size_t)NT * DD * 2);       // 2 MB
  float* mrow = (float*)(ws + (size_t)NT * DD * 4);                        // 16 KB
  float* srow = mrow + NT;                                                 // 16 KB

  k_convert<<<(NT * DD) / 256, 256, 0, stream>>>(feats, Xhi, Xlo, out);
  k_stats  <<<NT / 32, 256, 0, stream>>>(Xhi, Xlo, labels, mrow, srow);
  k_loss   <<<NT / 32, 256, 0, stream>>>(Xhi, Xlo, labels, mrow, srow, out);
}
